// Hierarchical_41231686042123
// MI455X (gfx1250) — compile-verified, matrix operand order checked by reading, not run on hardware
//
#include <hip/hip_runtime.h>

// CDNA5 / gfx1250 (wave32). WMMA f32<-f16 16x16x32 for the pairwise dot products.
typedef _Float16 v8h  __attribute__((ext_vector_type(8)));
typedef _Float16 v16h __attribute__((ext_vector_type(16)));
typedef float    v8f  __attribute__((ext_vector_type(8)));

constexpr int NPTS  = 65536;   // N
constexpr int DIM   = 8;       // D
constexpr int MCENT = 2048;    // M
constexpr int LLAY  = 4;       // L
constexpr int WAVES_PER_BLOCK = 8;   // 256 threads = 8 wave32
constexpr int ROWS_PER_WAVE   = 16;  // one 16-row tile of x per wave

// Raw v_exp_f32: the libm exp2f guard (cmp/cndmask/ldexp) only matters for
// denormal-magnitude results; our Gaussian underflows to 0 there anyway.
#if __has_builtin(__builtin_amdgcn_exp2f)
#define EXP2_RAW(x) __builtin_amdgcn_exp2f(x)
#else
__device__ __forceinline__ float exp2_raw_asm(float x) {
  float r;
  asm volatile("v_exp_f32 %0, %1\n\tv_nop\n\tv_nop" : "=v"(r) : "v"(x));
  return r;
}
#define EXP2_RAW(x) exp2_raw_asm(x)
#endif

union V16HU {
  v16h h;
  unsigned int u[8];
};

// ---------------------------------------------------------------------------
// Pre-pack x: f32 rows [N, 8] -> contiguous f16 rows (16B each, exactly the
// per-lane WMMA A/B operand layout for K=0..7) + per-row squared norm.
// ---------------------------------------------------------------------------
__global__ __launch_bounds__(256) void pack_x_kernel(
    const float* __restrict__ src, _Float16* __restrict__ dst,
    float* __restrict__ sq, int nrows) {
  int i = blockIdx.x * blockDim.x + threadIdx.x;
  if (i >= nrows) return;
  const float4* s4 = reinterpret_cast<const float4*>(src + (size_t)i * DIM);
  float4 lo = s4[0];
  float4 hi = s4[1];
  float ss = lo.x*lo.x + lo.y*lo.y + lo.z*lo.z + lo.w*lo.w
           + hi.x*hi.x + hi.y*hi.y + hi.z*hi.z + hi.w*hi.w;
  v8h h;
  h[0] = (_Float16)lo.x; h[1] = (_Float16)lo.y;
  h[2] = (_Float16)lo.z; h[3] = (_Float16)lo.w;
  h[4] = (_Float16)hi.x; h[5] = (_Float16)hi.y;
  h[6] = (_Float16)hi.z; h[7] = (_Float16)hi.w;
  *reinterpret_cast<v8h*>(dst + (size_t)i * DIM) = h;
  sq[i] = ss;
}

// ---------------------------------------------------------------------------
// Pre-pack centroids: f16 rows + interleaved {||mu||^2, w} float2 so the main
// loop needs a single b64 load for both per-centroid scalars.
// ---------------------------------------------------------------------------
__global__ __launch_bounds__(256) void pack_cent_kernel(
    const float* __restrict__ src, const float* __restrict__ coeffs,
    _Float16* __restrict__ dst, float2* __restrict__ m2w, int nrows) {
  int i = blockIdx.x * blockDim.x + threadIdx.x;
  if (i >= nrows) return;
  const float4* s4 = reinterpret_cast<const float4*>(src + (size_t)i * DIM);
  float4 lo = s4[0];
  float4 hi = s4[1];
  float ss = lo.x*lo.x + lo.y*lo.y + lo.z*lo.z + lo.w*lo.w
           + hi.x*hi.x + hi.y*hi.y + hi.z*hi.z + hi.w*hi.w;
  v8h h;
  h[0] = (_Float16)lo.x; h[1] = (_Float16)lo.y;
  h[2] = (_Float16)lo.z; h[3] = (_Float16)lo.w;
  h[4] = (_Float16)hi.x; h[5] = (_Float16)hi.y;
  h[6] = (_Float16)hi.z; h[7] = (_Float16)hi.w;
  *reinterpret_cast<v8h*>(dst + (size_t)i * DIM) = h;
  m2w[i] = make_float2(ss, coeffs[i]);
}

// ---------------------------------------------------------------------------
// Main kernel: each wave computes 16 rows of out. Loop layers x centroid
// tiles of 16; one v_wmma_f32_16x16x32_f16 per tile gives the 16x16 dot
// block; epilogue per element: ~3 VALU + 1 raw v_exp_f32.
// C layout: lane(0..15)=centroid n, VGPR r = x-row r (lanes 16..31: row r+8).
// ---------------------------------------------------------------------------
__global__ __launch_bounds__(WAVES_PER_BLOCK * 32) void gkm_main_kernel(
    const _Float16* __restrict__ xpk,  const float* __restrict__ x2,
    const _Float16* __restrict__ mupk, const float2* __restrict__ m2w,
    const float* __restrict__ widths,  float* __restrict__ out) {
  const int lane = threadIdx.x & 31;
  const int wave = threadIdx.x >> 5;
  const int n0   = (blockIdx.x * WAVES_PER_BLOCK + wave) * ROWS_PER_WAVE;

  // Loop-invariant lane mask: lanes 16..31 carry K=8..15/24..31 => all zero.
  const unsigned int lmask = (lane < 16) ? 0xFFFFFFFFu : 0u;

  // A operand: every lane loads row (n0 + (lane&15)); upper half masked to 0.
  V16HU av;
#pragma unroll
  for (int k = 0; k < 8; ++k) av.u[k] = 0u;
  {
    uint4 raw = *reinterpret_cast<const uint4*>(
        xpk + (size_t)(n0 + (lane & 15)) * DIM);
    av.u[0] = raw.x & lmask; av.u[1] = raw.y & lmask;
    av.u[2] = raw.z & lmask; av.u[3] = raw.w & lmask;
  }

  const int g = (lane >> 4) * 8;   // lanes 16..31 own rows n0+8..n0+15
  float x2v[8];
#pragma unroll
  for (int r = 0; r < 8; ++r) x2v[r] = x2[n0 + g + r];

  float outv[8];
#pragma unroll
  for (int r = 0; r < 8; ++r) outv[r] = 0.0f;

  constexpr float LOG2E     = 1.4426950408889634f;
  constexpr float TWO_PI_P4 = 1558.5454565440389f;  // (2*pi)^4

#pragma unroll
  for (int j = 0; j < LLAY; ++j) {
    const float sigma = widths[j];
    const float p2 = sigma * sigma;
    const float s2 = LOG2E / p2;                    // log2e / sigma^2
    const float p8 = (p2 * p2) * (p2 * p2);
    const float c2 = -__log2f(p8 * TWO_PI_P4);      // log2(gauss const)
    // K = 2^( s2*dot + (c2 - 0.5*s2*mu2) + (-0.5*s2*x2) )
    float negA[8];
#pragma unroll
    for (int r = 0; r < 8; ++r) negA[r] = -0.5f * s2 * x2v[r];

    float zacc[8], kacc[8];
#pragma unroll
    for (int r = 0; r < 8; ++r) { zacc[r] = 0.0f; kacc[r] = 0.0f; }

    const _Float16* mub = mupk + (size_t)j * MCENT * DIM;
    const float2*   mwb = m2w + j * MCENT;

    for (int m0 = 0; m0 < MCENT; m0 += 16) {
      const int mi = m0 + (lane & 15);   // this lane's centroid column
      // B operand: unconditional load + AND with lane mask (no EXEC churn).
      V16HU bv;
#pragma unroll
      for (int k = 4; k < 8; ++k) bv.u[k] = 0u;
      {
        uint4 raw = *reinterpret_cast<const uint4*>(mub + (size_t)mi * DIM);
        bv.u[0] = raw.x & lmask; bv.u[1] = raw.y & lmask;
        bv.u[2] = raw.z & lmask; bv.u[3] = raw.w & lmask;
      }
      const float2 mw = mwb[mi];                          // one b64 load
      const float  bc = __builtin_fmaf(-0.5f * s2, mw.x, c2);
      const float  wv = mw.y;

      v8f c = {};
      c = __builtin_amdgcn_wmma_f32_16x16x32_f16(
              /*neg_a=*/false, av.h, /*neg_b=*/false, bv.h,
              /*c_mod=*/(short)0, c, /*reuse_a=*/false, /*reuse_b=*/false);

#pragma unroll
      for (int r = 0; r < 8; ++r) {
        float arg = __builtin_fmaf(s2, c[r], bc + negA[r]);
        float K   = EXP2_RAW(arg);         // raw v_exp_f32
        zacc[r]  += K;
        kacc[r]   = __builtin_fmaf(K * K, wv, kacc[r]);
      }
    }

    // Reduce across the 16 lanes of each half-wave (masks 1,2,4,8 keep
    // lanes within their group of 16 in wave32).
#pragma unroll
    for (int mask = 1; mask <= 8; mask <<= 1) {
#pragma unroll
      for (int r = 0; r < 8; ++r) {
        zacc[r] += __shfl_xor(zacc[r], mask, 32);
        kacc[r] += __shfl_xor(kacc[r], mask, 32);
      }
    }
#pragma unroll
    for (int r = 0; r < 8; ++r)
      outv[r] += kacc[r] / (zacc[r] + 1e-10f);
  }

  if ((lane & 15) == 0) {
#pragma unroll
    for (int r = 0; r < 8; ++r) out[n0 + g + r] = outv[r];
  }
}

// ---------------------------------------------------------------------------
extern "C" void kernel_launch(void* const* d_in, const int* in_sizes, int n_in,
                              void* d_out, int out_size, void* d_ws, size_t ws_size,
                              hipStream_t stream) {
  (void)in_sizes; (void)n_in; (void)out_size; (void)ws_size;
  const float* x         = (const float*)d_in[0];   // [N, 8]
  const float* centroids = (const float*)d_in[1];   // [L, M, 8]
  const float* widths    = (const float*)d_in[2];   // [4]
  const float* coeffs    = (const float*)d_in[3];   // [L, M, 1]
  float* out = (float*)d_out;                       // [N, 1]

  // Workspace layout (~1.5 MB total):
  char* ws = (char*)d_ws;
  _Float16* xpk  = (_Float16*)(ws);                                  // 1 MiB
  float*    x2   = (float*)   (ws + 1048576);                        // 256 KiB
  _Float16* mupk = (_Float16*)(ws + 1048576 + 262144);               // 128 KiB
  float2*   m2w  = (float2*)  (ws + 1048576 + 262144 + 131072);      // 64 KiB

  pack_x_kernel<<<NPTS / 256, 256, 0, stream>>>(x, xpk, x2, NPTS);
  pack_cent_kernel<<<(LLAY * MCENT) / 256, 256, 0, stream>>>(
      centroids, coeffs, mupk, m2w, LLAY * MCENT);

  gkm_main_kernel<<<NPTS / (WAVES_PER_BLOCK * ROWS_PER_WAVE),
                    WAVES_PER_BLOCK * 32, 0, stream>>>(
      xpk, x2, mupk, m2w, widths, out);
}
